// DigitCapsule_10307921510899
// MI455X (gfx1250) — matrix operand order repaired and checked by reading, not compiled, not run on hardware
//
#include <hip/hip_runtime.h>
#include <hip/hip_bf16.h>

// Problem constants (match reference)
#define B_   32
#define R_   16384
#define C_   16
#define IC_  16
#define OC_  16

// Decomposition: one wave = one capsule c x RPW routes
#define RPW              64
#define WAVES_PER_BLOCK  8
#define WAVES_PER_C      (R_ / RPW)                          // 256
#define NBLOCKS          (C_ * WAVES_PER_C / WAVES_PER_BLOCK) // 512

// Workspace layout (floats):
//  accS   [0, 8192)                : num accumulator  (b*16+c)*16+o
//  accDen [8192, 8704)             : den accumulator  b*16+c
//  outBuf [8704, 16896)            : current routing outputs (b*16+c)*16+o
//  b1buf  [16896, 16896+16*16384*32): logits after iter 0, layout [c][r][b]
#define WS_ACCS   0
#define WS_ACCDEN 8192
#define WS_OUT    8704
#define WS_B1     16896

typedef __attribute__((ext_vector_type(16))) __bf16 v16bf;
typedef __attribute__((ext_vector_type(2)))  __bf16 v2bf;
typedef __attribute__((ext_vector_type(8)))  float  v8f;

union Frag {
  v16bf v;
  unsigned int w[8];
};

#if __has_builtin(__builtin_amdgcn_cvt_pk_bf16_f32)
// Packed convert if the toolchain exposes it (1 instr / pair).
__device__ __forceinline__ unsigned int pack2(float a, float b) {
  union { v2bf v; unsigned int u; } cv;
  cv.v = __builtin_amdgcn_cvt_pk_bf16_f32(a, b);
  return cv.u;
}
#else
// CDNA5 VOP3P mixed-precision FMA converts (ISA §15.10, opc 62/63):
//   v_fma_mixlo_bf16 D, S0, 1.0, 0  -> D[15:0]  = bf16(S0)
//   v_fma_mixhi_bf16 D, S0, 1.0, 0  -> D[31:16] = bf16(S0), D[15:0] preserved
// 2 instructions per packed pair, no half-register shuffling.
__device__ __forceinline__ unsigned int pack2(float a, float b) {
  unsigned int r;
  asm("v_fma_mixlo_bf16 %0, %1, 1.0, 0" : "=v"(r) : "v"(a));
  asm("v_fma_mixhi_bf16 %0, %1, 1.0, 0" : "+v"(r) : "v"(b));
  return r;
}
#endif

// ---------------------------------------------------------------------------
// zero the small accumulators once per launch
__global__ void k_zero(float* __restrict__ ws) {
  int t = blockIdx.x * blockDim.x + threadIdx.x;
  if (t < WS_OUT) ws[t] = 0.0f;
}

// ---------------------------------------------------------------------------
// Fused W-streaming pass. Recomputes u_hat[o,b] tiles with WMMA and reduces.
// MODE 0: accS += sum_r u_hat
// MODE 1: b1 = u_hat . out0 ; store b1 ; accS += e^{b1} u_hat ; accDen += e^{b1}
// MODE 2: b2 = b1 + u_hat . out1 ;       accS += e^{b2} u_hat ; accDen += e^{b2}
template <int MODE>
__global__ __launch_bounds__(256, 2)
void k_route(const float* __restrict__ x, const float* __restrict__ W,
             float* __restrict__ ws) {
  float*       accS   = ws + WS_ACCS;
  float*       accDen = ws + WS_ACCDEN;
  const float* outBuf = ws + WS_OUT;
  float*       b1buf  = ws + WS_B1;

  __shared__ float sm[544];                 // 512 num + 32 den, per-block (one c)
  for (int t = threadIdx.x; t < 544; t += blockDim.x) sm[t] = 0.0f;
  __syncthreads();

  const int lane = threadIdx.x & 31;
  const int row  = lane & 15;               // A: M row (=o); B/D: column (=b%16)
  const int hi   = lane >> 4;               // lane half
  const int w    = blockIdx.x * WAVES_PER_BLOCK + (threadIdx.x >> 5);
  const int c    = w >> 8;                  // w / WAVES_PER_C  (block-uniform)
  const int rbase = (w & (WAVES_PER_C - 1)) * RPW;

  // ---- B fragments (x for capsule c), two batch halves, resident in VGPRs.
  // B 32x16 bf16 layout: column N=lane%16; lane<16 -> K0..7 = i 0..7,
  // lane>=16 -> i 8..15; K 16..31 zero-padded.
  Frag bf0, bf1;
#pragma unroll
  for (int k = 0; k < 8; k++) { bf0.w[k] = 0u; bf1.w[k] = 0u; }
  {
    const float* xp0 = x + ((size_t)(row)      * C_ + c) * IC_ + 8 * hi;
    const float* xp1 = x + ((size_t)(row + 16) * C_ + c) * IC_ + 8 * hi;
#pragma unroll
    for (int k = 0; k < 4; k++) {
      bf0.w[k] = pack2(xp0[2 * k], xp0[2 * k + 1]);
      bf1.w[k] = pack2(xp1[2 * k], xp1[2 * k + 1]);
    }
  }

  // ---- routing outputs for the dot product (modes 1,2); this lane's 8 o's
  float outv0[8], outv1[8];
#pragma unroll
  for (int v = 0; v < 8; v++) { outv0[v] = 0.0f; outv1[v] = 0.0f; }
  if (MODE >= 1) {
#pragma unroll
    for (int v = 0; v < 8; v++) {
      const int o = v + 8 * hi;
      outv0[v] = outBuf[((size_t)(row)      * C_ + c) * OC_ + o];
      outv1[v] = outBuf[((size_t)(row + 16) * C_ + c) * OC_ + o];
    }
  }

  float acc0[8], acc1[8];
#pragma unroll
  for (int v = 0; v < 8; v++) { acc0[v] = 0.0f; acc1[v] = 0.0f; }
  float den0 = 0.0f, den1 = 0.0f;

  // A tile for (r,c): W[r][c][0..15][0..15], 1 KB fp32 contiguous.
  // A 16x32 bf16 layout: lane<16 -> M=row, K0..7 = i 0..7 ; lane>=16 -> i 8..15;
  // K 16..31 zero-padded.
  const float* aptr = W + ((size_t)(rbase * C_ + c)) * (OC_ * IC_)
                        + row * IC_ + 8 * hi;

#pragma unroll 2
  for (int rr = 0; rr < RPW; rr++) {
    const int r = rbase + rr;
    const float4 a0 = *(const float4*)(aptr);
    const float4 a1 = *(const float4*)(aptr + 4);
    // prefetch next route's tile ahead of the consuming loads
    if (rr + 1 < RPW) __builtin_prefetch(aptr + C_ * OC_ * IC_, 0, 3);

    Frag af;
    af.w[0] = pack2(a0.x, a0.y);
    af.w[1] = pack2(a0.z, a0.w);
    af.w[2] = pack2(a1.x, a1.y);
    af.w[3] = pack2(a1.z, a1.w);
    af.w[4] = af.w[5] = af.w[6] = af.w[7] = 0u;   // K=16..31 zero pad

    v8f z = {0.f, 0.f, 0.f, 0.f, 0.f, 0.f, 0.f, 0.f};
    // D[o,b] tiles: u_hat for batch halves 0 and 1
    v8f d0 = __builtin_amdgcn_wmma_f32_16x16x32_bf16(
        false, af.v, false, bf0.v, (short)0, z, false, false);
    v8f d1 = __builtin_amdgcn_wmma_f32_16x16x32_bf16(
        false, af.v, false, bf1.v, (short)0, z, false, false);

    if (MODE == 0) {
#pragma unroll
      for (int v = 0; v < 8; v++) { acc0[v] += d0[v]; acc1[v] += d1[v]; }
    } else {
      // logit contribution: sum over o (pair lanes L, L^16 cover all 16 o's)
      float dot0 = 0.0f, dot1 = 0.0f;
#pragma unroll
      for (int v = 0; v < 8; v++) {
        dot0 += d0[v] * outv0[v];
        dot1 += d1[v] * outv1[v];
      }
      dot0 += __shfl_xor(dot0, 16, 32);
      dot1 += __shfl_xor(dot1, 16, 32);

      float bv0 = dot0, bv1 = dot1;
      float* bp = b1buf + ((size_t)c * R_ + r) * B_;
      if (MODE == 2) {                 // b2 = b1 + u_hat.out1
        bv0 += bp[row];
        bv1 += bp[row + 16];
      }
      if (MODE == 1 && hi == 0) {      // persist b1 (coalesced 128B per (c,r))
        bp[row]      = bv0;
        bp[row + 16] = bv1;
      }
      const float e0 = __expf(bv0);
      const float e1 = __expf(bv1);
#pragma unroll
      for (int v = 0; v < 8; v++) {
        acc0[v] += e0 * d0[v];
        acc1[v] += e1 * d1[v];
      }
      if (hi == 0) { den0 += e0; den1 += e1; }
    }
    aptr += C_ * OC_ * IC_;            // next route
  }

  // ---- block-level reduction in LDS (all waves in block share c)
#pragma unroll
  for (int v = 0; v < 8; v++) {
    const int o = v + 8 * hi;
    atomicAdd(&sm[(row)      * 16 + o], acc0[v]);
    atomicAdd(&sm[(row + 16) * 16 + o], acc1[v]);
  }
  if (MODE >= 1 && hi == 0) {
    atomicAdd(&sm[512 + row],      den0);
    atomicAdd(&sm[512 + 16 + row], den1);
  }
  __syncthreads();

  for (int t = threadIdx.x; t < 544; t += blockDim.x) {
    if (t < 512) {
      const int b = t >> 4, o = t & 15;
      atomicAdd(&accS[((size_t)b * C_ + c) * OC_ + o], sm[t]);
    } else if (MODE >= 1) {
      const int b = t - 512;
      atomicAdd(&accDen[b * C_ + c], sm[t]);
    }
  }
}

// ---------------------------------------------------------------------------
// s = accS/den (or /R for iter 0); outputs = squash(s); re-zero accumulators.
// MODE 0/1 -> write outBuf; MODE 2 -> write final d_out (B,C,OC).
template <int MODE>
__global__ void k_squash(float* __restrict__ ws, float* __restrict__ dout) {
  const int t = threadIdx.x;               // 512 threads: t = b*16+c
  float* accS   = ws + WS_ACCS;
  float* accDen = ws + WS_ACCDEN;
  float* outBuf = ws + WS_OUT;

  const float inv = (MODE == 0) ? (1.0f / (float)R_) : (1.0f / accDen[t]);
  float s[16];
  float nrm2 = 0.0f;
#pragma unroll
  for (int o = 0; o < 16; o++) {
    const float v = accS[t * 16 + o] * inv;
    s[o] = v;
    nrm2 += v * v;
  }
  const float nrm   = sqrtf(nrm2);
  const float scale = nrm / (1.0f + nrm2);
#pragma unroll
  for (int o = 0; o < 16; o++) {
    const float v = scale * s[o];
    if (MODE == 2) dout[t * 16 + o]  = v;   // (B,C,OC) row-major == t*16+o
    else           outBuf[t * 16 + o] = v;
  }
  if (MODE != 2) {                          // reset accumulators for next pass
#pragma unroll
    for (int o = 0; o < 16; o++) accS[t * 16 + o] = 0.0f;
    accDen[t] = 0.0f;
  }
}

// ---------------------------------------------------------------------------
extern "C" void kernel_launch(void* const* d_in, const int* in_sizes, int n_in,
                              void* d_out, int out_size, void* d_ws, size_t ws_size,
                              hipStream_t stream) {
  const float* x = (const float*)d_in[0];   // (B, C, IC)
  const float* W = (const float*)d_in[1];   // (R, C, OC, IC)
  float* out = (float*)d_out;               // (B, C, OC)
  float* ws  = (float*)d_ws;                // needs ~33.6 MB

  k_zero<<<WS_OUT / 256, 256, 0, stream>>>(ws);

  // iter 0: uniform coupling -> s0 = (1/R) sum_r u_hat
  k_route<0><<<NBLOCKS, 256, 0, stream>>>(x, W, ws);
  k_squash<0><<<1, 512, 0, stream>>>(ws, out);

  // iter 1: b1 = u_hat.out0 ; softmax-weighted sum (unnormalized num/den)
  k_route<1><<<NBLOCKS, 256, 0, stream>>>(x, W, ws);
  k_squash<1><<<1, 512, 0, stream>>>(ws, out);

  // iter 2: b2 = b1 + u_hat.out1 ; final outputs
  k_route<2><<<NBLOCKS, 256, 0, stream>>>(x, W, ws);
  k_squash<2><<<1, 512, 0, stream>>>(ws, out);
}
